// QwenVLImageEncoder_23433341567855
// MI455X (gfx1250) — matrix operand order repaired and checked by reading, not compiled, not run on hardware
//
#include <hip/hip_runtime.h>
#include <math.h>
#include <stdint.h>

typedef __bf16 bf16;
typedef bf16  v16bf   __attribute__((ext_vector_type(16)));
typedef bf16  bf16x4  __attribute__((ext_vector_type(4)));
typedef float v8f     __attribute__((ext_vector_type(8)));
typedef float float4v __attribute__((ext_vector_type(4)));

#define S_TOK 2304
#define D_MODEL 1280
#define NH 16
#define HD 80
#define HDP 96   // head dim padded to 3*32 for WMMA K
#define FF 5120
#define MU 4
#define MERGE_DIM 5120
#define OUT_DIM 3584
#define SROWS 576

__device__ __forceinline__ float gelu_f(float x) {
  float x3 = x * x * x;
  return 0.5f * x * (1.f + tanhf(0.7978845608028654f * (x + 0.044715f * x3)));
}

// ---- CDNA5 async global->LDS (ASYNCcnt path) ------------------------------
__device__ __forceinline__ void async_copy_b128(uint32_t lds_off, const void* g) {
  asm volatile("global_load_async_to_lds_b128 %0, %1, off"
               :: "v"(lds_off), "v"(g) : "memory");
}
__device__ __forceinline__ void wait_async0() {
  asm volatile("s_wait_asynccnt 0" ::: "memory");
}

// ---------------------------------------------------------------------------
// Weight transpose + fp32->bf16 convert: W (KxN f32) -> Wt (NxK bf16)
// ---------------------------------------------------------------------------
__global__ __launch_bounds__(256) void transpose_cvt_kernel(
    const float* __restrict__ W, bf16* __restrict__ Wt, int K, int N) {
  __shared__ float t[32][33];
  int kb = blockIdx.x * 32, nb = blockIdx.y * 32;
  int tx = threadIdx.x & 31, ty = threadIdx.x >> 5;  // 32x8
#pragma unroll
  for (int r = ty; r < 32; r += 8)
    t[r][tx] = W[(size_t)(kb + r) * N + nb + tx];
  __syncthreads();
#pragma unroll
  for (int r = ty; r < 32; r += 8)
    Wt[(size_t)(nb + r) * K + kb + tx] = (bf16)t[tx][r];
}

// ---------------------------------------------------------------------------
// Main WMMA GEMM, bf16 operands, async-to-LDS double buffering.
// A: bf16 MxK row-major. Bt: bf16 NxK row-major (pre-transposed weight).
// C: fp32 (act(A@B + bias) [+ resid]) or bf16 (OUTBF16).
// 256 threads / 8 waves; tile 128x128; K-step 32.
// ---------------------------------------------------------------------------
template <int OUTBF16>
__global__ __launch_bounds__(256) void gemm_bf16_async(
    const bf16* __restrict__ A, const bf16* __restrict__ Bt,
    const float* __restrict__ bias, const float* __restrict__ resid,
    void* __restrict__ Cout, int M, int N, int K, int act_gelu) {
  __shared__ bf16 lA[2][128 * 40];
  __shared__ bf16 lB[2][128 * 40];
  const int tid = threadIdx.x;
  const int wave = tid >> 5, lane = tid & 31;
  const int half = lane >> 4, nl = lane & 15;
  const int n0 = blockIdx.x * 128, m0 = blockIdx.y * 128;

  // 512 16-byte slots per 128x32 bf16 tile -> 2 async B128 per thread per tile
  auto issue_tiles = [&](int bi, int k0) {
#pragma unroll
    for (int i = 0; i < 2; i++) {
      int p = tid + 256 * i;
      int r = p >> 2, q = p & 3;  // row, 16B chunk (8 bf16)
      {
        int gm = min(m0 + r, M - 1);
        async_copy_b128((uint32_t)(uintptr_t)&lA[bi][r * 40 + q * 8],
                        A + (size_t)gm * K + k0 + q * 8);
      }
      {
        int gn = min(n0 + r, N - 1);
        async_copy_b128((uint32_t)(uintptr_t)&lB[bi][r * 40 + q * 8],
                        Bt + (size_t)gn * K + k0 + q * 8);
      }
    }
  };

  v8f z = {0.f, 0.f, 0.f, 0.f, 0.f, 0.f, 0.f, 0.f};
  v8f acc[8] = {z, z, z, z, z, z, z, z};

  issue_tiles(0, 0);
  wait_async0();
  __syncthreads();

  int buf = 0, k0 = 0;
  while (true) {
    int kn = k0 + 32;
    bool more = kn < K;
    if (more) {
      issue_tiles(buf ^ 1, kn);  // async engine fills other buffer
      if (kn + 32 < K)           // global_prefetch_b8 one tile further
        __builtin_prefetch(A + (size_t)min(m0 + (tid >> 1), M - 1) * K + kn + 32, 0, 1);
    }
    v16bf af;
#pragma unroll
    for (int e = 0; e < 16; e++) {
      int k = (e & 7) + half * 8 + ((e >> 3) << 4);
      af[e] = lA[buf][(wave * 16 + nl) * 40 + k];
    }
#pragma unroll
    for (int nc = 0; nc < 8; nc++) {
      v16bf bfrag;
#pragma unroll
      for (int e = 0; e < 16; e++)
        bfrag[e] = lB[buf][(nc * 16 + nl) * 40 + (e & 7) + half * 8 + ((e >> 3) << 4)];
      acc[nc] = __builtin_amdgcn_wmma_f32_16x16x32_bf16(
          false, af, false, bfrag, (short)0, acc[nc], false, false);
    }
    if (!more) break;
    wait_async0();
    __syncthreads();
    buf ^= 1;
    k0 = kn;
  }

#pragma unroll
  for (int nc = 0; nc < 8; nc++) {
    int gn = n0 + nc * 16 + nl;
    if (gn >= N) continue;
    float bv = bias ? bias[gn] : 0.f;
#pragma unroll
    for (int j = 0; j < 8; j++) {
      int gm = m0 + wave * 16 + j + 8 * half;
      if (gm >= M) continue;
      float v = acc[nc][j] + bv;
      if (act_gelu) v = gelu_f(v);
      if (OUTBF16) {
        ((bf16*)Cout)[(size_t)gm * N + gn] = (bf16)v;
      } else {
        if (resid) v += resid[(size_t)gm * N + gn];
        ((float*)Cout)[(size_t)gm * N + gn] = v;
      }
    }
  }
}

// ---------------------------------------------------------------------------
// fp32-source GEMM for patch embed only: A f32 MxK, B f32 NxK (transposed use),
// ragged K; register-staged double buffering.
// ---------------------------------------------------------------------------
__global__ __launch_bounds__(256) void gemm_f32_patch(
    const float* __restrict__ A, const float* __restrict__ B,
    const float* __restrict__ bias, float* __restrict__ C,
    int M, int N, int K) {
  __shared__ bf16 lA[2][128 * 40];
  __shared__ bf16 lB[2][128 * 40];
  const int tid = threadIdx.x;
  const int wave = tid >> 5, lane = tid & 31;
  const int half = lane >> 4, nl = lane & 15;
  const int n0 = blockIdx.x * 128, m0 = blockIdx.y * 128;

  float4v ra[4], rb[4];
  auto load_tiles = [&](int k0) {
#pragma unroll
    for (int i = 0; i < 4; i++) {
      int p = tid + 256 * i;
      int r = p >> 3, kq = p & 7;
      int gm = min(m0 + r, M - 1);
      int gn = min(n0 + r, N - 1);
#pragma unroll
      for (int e = 0; e < 4; e++) {
        int kk = k0 + kq * 4 + e;
        ra[i][e] = (kk < K) ? A[(size_t)gm * K + kk] : 0.f;
        rb[i][e] = (kk < K) ? B[(size_t)gn * K + kk] : 0.f;
      }
    }
  };
  auto store_tiles = [&](int bi) {
#pragma unroll
    for (int i = 0; i < 4; i++) {
      int p = tid + 256 * i;
      int r = p >> 3, kq = p & 7;
      bf16x4 ba, bb;
#pragma unroll
      for (int e = 0; e < 4; e++) { ba[e] = (bf16)ra[i][e]; bb[e] = (bf16)rb[i][e]; }
      *(bf16x4*)&lA[bi][r * 40 + kq * 4] = ba;
      *(bf16x4*)&lB[bi][r * 40 + kq * 4] = bb;
    }
  };

  v8f z = {0.f, 0.f, 0.f, 0.f, 0.f, 0.f, 0.f, 0.f};
  v8f acc[8] = {z, z, z, z, z, z, z, z};
  load_tiles(0);
  store_tiles(0);
  __syncthreads();
  int buf = 0, k0 = 0;
  while (true) {
    int kn = k0 + 32;
    bool more = kn < K;
    if (more) load_tiles(kn);
    v16bf af;
#pragma unroll
    for (int e = 0; e < 16; e++) {
      int k = (e & 7) + half * 8 + ((e >> 3) << 4);
      af[e] = lA[buf][(wave * 16 + nl) * 40 + k];
    }
#pragma unroll
    for (int nc = 0; nc < 8; nc++) {
      v16bf bfrag;
#pragma unroll
      for (int e = 0; e < 16; e++)
        bfrag[e] = lB[buf][(nc * 16 + nl) * 40 + (e & 7) + half * 8 + ((e >> 3) << 4)];
      acc[nc] = __builtin_amdgcn_wmma_f32_16x16x32_bf16(
          false, af, false, bfrag, (short)0, acc[nc], false, false);
    }
    if (!more) break;
    store_tiles(buf ^ 1);
    __syncthreads();
    buf ^= 1;
    k0 = kn;
  }
#pragma unroll
  for (int nc = 0; nc < 8; nc++) {
    int gn = n0 + nc * 16 + nl;
    if (gn >= N) continue;
    float bv = bias ? bias[gn] : 0.f;
#pragma unroll
    for (int j = 0; j < 8; j++) {
      int gm = m0 + wave * 16 + j + 8 * half;
      if (gm >= M) continue;
      C[(size_t)gm * N + gn] = acc[nc][j] + bv;
    }
  }
}

// ---------------------------------------------------------------------------
// RMSNorm: one block per row, bf16 output
// ---------------------------------------------------------------------------
__global__ __launch_bounds__(256) void rmsnorm_kernel(
    const float* __restrict__ x, const float* __restrict__ g,
    bf16* __restrict__ out, int N) {
  __shared__ float red[256];
  int row = blockIdx.x, tid = threadIdx.x;
  const float* xr = x + (size_t)row * N;
  float s = 0.f;
  for (int i = tid; i < N; i += 256) { float v = xr[i]; s += v * v; }
  red[tid] = s; __syncthreads();
  for (int off = 128; off > 0; off >>= 1) {
    if (tid < off) red[tid] += red[tid + off];
    __syncthreads();
  }
  float r = rsqrtf(red[0] / (float)N + 1e-6f);
  for (int i = tid; i < N; i += 256)
    out[(size_t)row * N + i] = (bf16)(xr[i] * r * g[i]);
}

// ---------------------------------------------------------------------------
// Per-row window range from dense mask (0.0 / -1e9 block diagonal)
// ---------------------------------------------------------------------------
__global__ __launch_bounds__(256) void mask_range_kernel(
    const float* __restrict__ mask, int* __restrict__ rs, int* __restrict__ re) {
  __shared__ int smin[256], smax[256];
  int row = blockIdx.x, tid = threadIdx.x;
  int lo = S_TOK, hi = 0;
  const float* mr = mask + (size_t)row * S_TOK;
  for (int j = tid; j < S_TOK; j += 256)
    if (mr[j] == 0.f) { lo = min(lo, j); hi = max(hi, j + 1); }
  smin[tid] = lo; smax[tid] = hi; __syncthreads();
  for (int off = 128; off > 0; off >>= 1) {
    if (tid < off) { smin[tid] = min(smin[tid], smin[tid + off]);
                     smax[tid] = max(smax[tid], smax[tid + off]); }
    __syncthreads();
  }
  if (tid == 0) { rs[row] = smin[0]; re[row] = smax[0]; }
}

// ---------------------------------------------------------------------------
// RoPE + repack qkv (f32) into bf16 padded per-head buffers (S,NH,96)
// ---------------------------------------------------------------------------
__global__ __launch_bounds__(96) void rope_pad_kernel(
    const float* __restrict__ qkv, const float* __restrict__ cosb,
    const float* __restrict__ sinb, bf16* __restrict__ qp,
    bf16* __restrict__ kp, bf16* __restrict__ vp) {
  int s = blockIdx.x, h = blockIdx.y, d = threadIdx.x;
  const float* q = qkv + (size_t)s * (3 * D_MODEL) + h * HD;
  const float* k = q + D_MODEL;
  const float* v = q + 2 * D_MODEL;
  float qo = 0.f, ko = 0.f, vo = 0.f;
  if (d < HD) {
    if (d < 40) {
      float c = cosb[s * 40 + d], sn = sinb[s * 40 + d];
      qo = q[d] * c - q[d + 40] * sn;
      ko = k[d] * c - k[d + 40] * sn;
    } else {
      int i = d - 40;
      float c = cosb[s * 40 + i], sn = sinb[s * 40 + i];
      qo = q[d] * c + q[i] * sn;
      ko = k[d] * c + k[i] * sn;
    }
    vo = v[d];
  }
  size_t base = ((size_t)s * NH + h) * HDP;
  qp[base + d] = (bf16)qo;
  kp[base + d] = (bf16)ko;
  vp[base + d] = (bf16)vo;
}

// ---------------------------------------------------------------------------
// Flash attention: grid (S/64, NH), 128 threads (4 waves x 16 query rows).
// K/V tiles async-copied to LDS (bf16 -> bf16), bf16 output.
// ---------------------------------------------------------------------------
__global__ __launch_bounds__(128) void flash_attn_kernel(
    const bf16* __restrict__ qp, const bf16* __restrict__ kp,
    const bf16* __restrict__ vp, const int* __restrict__ rs,
    const int* __restrict__ re, const unsigned char* __restrict__ flags,
    int layer, bf16* __restrict__ out) {
  __shared__ bf16 lK[32 * 104];     // [key][hdp]
  __shared__ bf16 lV[32 * 104];     // [key][hdp] (cols >=80 are zero pad)
  __shared__ bf16 lP[4 * 16 * 40];  // per-wave P tiles [m][key]
  const float scale = 0.11180339887498948f;  // 1/sqrt(80)
  const int tid = threadIdx.x;
  const int wave = tid >> 5, lane = tid & 31;
  const int half = lane >> 4, nl = lane & 15;
  const int h = blockIdx.y;
  const int qb = blockIdx.x * 64;
  const int qrow = qb + wave * 16 + nl;

  v16bf aq[3];
  const bf16* qptr = qp + ((size_t)qrow * NH + h) * HDP;
#pragma unroll
  for (int c = 0; c < 3; c++) {
#pragma unroll
    for (int e = 0; e < 16; e++) {
      int kd = c * 32 + (e & 7) + half * 8 + ((e >> 3) << 4);
      aq[c][e] = (bf16)((float)qptr[kd] * scale);
    }
  }

  int start, end;
  if (flags[layer]) { start = 0; end = S_TOK; }
  else              { start = rs[qb]; end = re[qb]; }

  v8f z = {0.f, 0.f, 0.f, 0.f, 0.f, 0.f, 0.f, 0.f};
  v8f acc[5] = {z, z, z, z, z};
  float mrow[8], lrow[8];
#pragma unroll
  for (int j = 0; j < 8; j++) { mrow[j] = -1e30f; lrow[j] = 0.f; }

  for (int kt = start; kt < end; kt += 32) {
    __syncthreads();  // all waves done with previous lK/lV before refill
    // K & V tiles: 32 rows x 12 chunks of 16B -> 3 async B128 each per thread
#pragma unroll
    for (int i = 0; i < 3; i++) {
      int p = tid + 128 * i;
      int key = p / 12, q = p % 12;
      size_t gb = ((size_t)(kt + key) * NH + h) * HDP + q * 8;
      async_copy_b128((uint32_t)(uintptr_t)&lK[key * 104 + q * 8], kp + gb);
      async_copy_b128((uint32_t)(uintptr_t)&lV[key * 104 + q * 8], vp + gb);
    }
    wait_async0();
    __syncthreads();

    // Scores S[16 x 32] = Q @ K^T (two 16-key halves, 3 K-chunks each)
    v8f st[2];
#pragma unroll
    for (int n2 = 0; n2 < 2; n2++) {
      v8f sc = z;
#pragma unroll
      for (int c = 0; c < 3; c++) {
        v16bf b;
#pragma unroll
        for (int e = 0; e < 16; e++)
          b[e] = lK[(n2 * 16 + nl) * 104 + c * 32 + (e & 7) + half * 8 + ((e >> 3) << 4)];
        sc = __builtin_amdgcn_wmma_f32_16x16x32_bf16(
            false, aq[c], false, b, (short)0, sc, false, false);
      }
      st[n2] = sc;
    }

    // Online softmax: row M = j + 8*half lives on one 16-lane group
#pragma unroll
    for (int j = 0; j < 8; j++) {
      float v = fmaxf(st[0][j], st[1][j]);
#pragma unroll
      for (int off = 8; off > 0; off >>= 1) v = fmaxf(v, __shfl_xor(v, off, 16));
      float mn = fmaxf(mrow[j], v);
      float alpha = __expf(mrow[j] - mn);
      mrow[j] = mn;
      float p0 = __expf(st[0][j] - mn);
      float p1 = __expf(st[1][j] - mn);
      float ps = p0 + p1;
#pragma unroll
      for (int off = 8; off > 0; off >>= 1) ps += __shfl_xor(ps, off, 16);
      lrow[j] = lrow[j] * alpha + ps;
#pragma unroll
      for (int nc = 0; nc < 5; nc++) acc[nc][j] *= alpha;
      lP[wave * 640 + (j + 8 * half) * 40 + nl] = (bf16)p0;
      lP[wave * 640 + (j + 8 * half) * 40 + 16 + nl] = (bf16)p1;
    }
    __syncthreads();

    // P as A fragment (16x32), V as B (32 x 80 in 5 chunks)
    v16bf pa;
#pragma unroll
    for (int e = 0; e < 16; e++) {
      int k = (e & 7) + half * 8 + ((e >> 3) << 4);
      pa[e] = lP[wave * 640 + nl * 40 + k];
    }
#pragma unroll
    for (int nc = 0; nc < 5; nc++) {
      v16bf b;
#pragma unroll
      for (int e = 0; e < 16; e++)
        b[e] = lV[((e & 7) + half * 8 + ((e >> 3) << 4)) * 104 + nc * 16 + nl];
      acc[nc] = __builtin_amdgcn_wmma_f32_16x16x32_bf16(
          false, pa, false, b, (short)0, acc[nc], false, false);
    }
  }

#pragma unroll
  for (int nc = 0; nc < 5; nc++) {
#pragma unroll
    for (int j = 0; j < 8; j++) {
      int gm = qb + wave * 16 + j + 8 * half;
      int gn = nc * 16 + nl;
      out[(size_t)gm * D_MODEL + h * HD + gn] = (bf16)(acc[nc][j] / lrow[j]);
    }
  }
}

// ---------------------------------------------------------------------------
// Row permutations
// ---------------------------------------------------------------------------
__global__ __launch_bounds__(256) void permute_groups_kernel(
    const float* __restrict__ in, const int* __restrict__ wi,
    float* __restrict__ out, int Dm) {
  int r = blockIdx.x;
  int g = r / MU, j = r % MU;
  int src = wi[g] * MU + j;
  for (int d = threadIdx.x; d < Dm; d += 256)
    out[(size_t)r * Dm + d] = in[(size_t)src * Dm + d];
}

__global__ __launch_bounds__(256) void gather_rows_kernel(
    const float* __restrict__ in, const int* __restrict__ idx,
    float* __restrict__ out, int Dm) {
  int r = blockIdx.x;
  int src = idx[r];
  for (int d = threadIdx.x; d < Dm; d += 256)
    out[(size_t)r * Dm + d] = in[(size_t)src * Dm + d];
}

// ---------------------------------------------------------------------------
// Host orchestration
// ---------------------------------------------------------------------------
extern "C" void kernel_launch(void* const* d_in, const int* in_sizes, int n_in,
                              void* d_out, int out_size, void* d_ws, size_t ws_size,
                              hipStream_t stream) {
  (void)in_sizes; (void)n_in; (void)out_size; (void)ws_size;
  const float* pixel   = (const float*)d_in[0];
  const float* patch_w = (const float*)d_in[1];
  const float* patch_b = (const float*)d_in[2];
  const float* w_qkv   = (const float*)d_in[3];
  const float* b_qkv   = (const float*)d_in[4];
  const float* w_o     = (const float*)d_in[5];
  const float* b_o     = (const float*)d_in[6];
  const float* g1      = (const float*)d_in[7];
  const float* g2      = (const float*)d_in[8];
  const float* w_up    = (const float*)d_in[9];
  const float* b_up    = (const float*)d_in[10];
  const float* w_down  = (const float*)d_in[11];
  const float* b_down  = (const float*)d_in[12];
  const float* final_g = (const float*)d_in[13];
  const float* m_w1    = (const float*)d_in[14];
  const float* m_b1    = (const float*)d_in[15];
  const float* m_w2    = (const float*)d_in[16];
  const float* m_b2    = (const float*)d_in[17];
  const int*   wi      = (const int*)d_in[18];
  const int*   rev     = (const int*)d_in[19];
  const float* mask    = (const float*)d_in[20];
  const float* rcos    = (const float*)d_in[21];
  const float* rsin    = (const float*)d_in[22];
  const unsigned char* flags = (const unsigned char*)d_in[23];
  float* outp = (float*)d_out;

  char* ws = (char*)d_ws;
  size_t off = 0;
  auto allocf = [&](size_t n) -> float* {
    float* p = (float*)(ws + off);
    off += ((n * 4 + 255) / 256) * 256;
    return p;
  };
  auto allocb = [&](size_t n) -> bf16* {
    bf16* p = (bf16*)(ws + off);
    off += ((n * 2 + 255) / 256) * 256;
    return p;
  };
  // fp32 activations
  float* tmp = allocf((size_t)S_TOK * D_MODEL);
  float* x   = allocf((size_t)S_TOK * D_MODEL);
  float* qkv = allocf((size_t)S_TOK * 3 * D_MODEL);
  float* y   = allocf((size_t)SROWS * OUT_DIM);
  // bf16 activations
  bf16* h_bf  = allocb((size_t)S_TOK * D_MODEL);   // also 576x5120 view
  bf16* a_bf  = allocb((size_t)S_TOK * D_MODEL);   // attention output
  bf16* ff_bf = allocb((size_t)S_TOK * FF);        // also merge hidden
  bf16* qp = allocb((size_t)S_TOK * NH * HDP);
  bf16* kp = allocb((size_t)S_TOK * NH * HDP);
  bf16* vp = allocb((size_t)S_TOK * NH * HDP);
  // bf16 pre-transposed weights (NxK)
  bf16* wqkvT = allocb((size_t)8 * 3 * D_MODEL * D_MODEL);
  bf16* woT   = allocb((size_t)8 * D_MODEL * D_MODEL);
  bf16* wupT  = allocb((size_t)8 * FF * D_MODEL);
  bf16* wdnT  = allocb((size_t)8 * D_MODEL * FF);
  bf16* mw1T  = allocb((size_t)MERGE_DIM * MERGE_DIM);
  bf16* mw2T  = allocb((size_t)OUT_DIM * MERGE_DIM);
  int* rowS = (int*)(ws + off); off += ((S_TOK * 4 + 255) / 256) * 256;
  int* rowE = (int*)(ws + off); off += ((S_TOK * 4 + 255) / 256) * 256;

  auto tcvt = [&](const float* W, bf16* Wt, int K, int N) {
    transpose_cvt_kernel<<<dim3(K / 32, N / 32), 256, 0, stream>>>(W, Wt, K, N);
  };
  // One-time weight conversion (bf16, transposed to NxK)
  for (int l = 0; l < 8; l++) {
    tcvt(w_qkv  + (size_t)l * D_MODEL * 3 * D_MODEL, wqkvT + (size_t)l * 3 * D_MODEL * D_MODEL, D_MODEL, 3 * D_MODEL);
    tcvt(w_o    + (size_t)l * D_MODEL * D_MODEL,     woT   + (size_t)l * D_MODEL * D_MODEL,     D_MODEL, D_MODEL);
    tcvt(w_up   + (size_t)l * D_MODEL * FF,          wupT  + (size_t)l * FF * D_MODEL,          D_MODEL, FF);
    tcvt(w_down + (size_t)l * FF * D_MODEL,          wdnT  + (size_t)l * D_MODEL * FF,          FF, D_MODEL);
  }
  tcvt(m_w1, mw1T, MERGE_DIM, MERGE_DIM);
  tcvt(m_w2, mw2T, MERGE_DIM, OUT_DIM);

  mask_range_kernel<<<S_TOK, 256, 0, stream>>>(mask, rowS, rowE);

  dim3 blk(256);
  auto gblk = [](int N, int M) { return dim3(N / 128, (M + 127) / 128); };

  // patch embed (fp32, ragged K, patch_w already NxK) -> tmp -> permute -> x
  gemm_f32_patch<<<gblk(D_MODEL, S_TOK), blk, 0, stream>>>(
      pixel, patch_w, patch_b, tmp, S_TOK, D_MODEL, 1176);
  permute_groups_kernel<<<S_TOK, 256, 0, stream>>>(tmp, wi, x, D_MODEL);

  for (int l = 0; l < 8; l++) {
    const bf16* wqkv_l = wqkvT + (size_t)l * 3 * D_MODEL * D_MODEL;
    const bf16* wo_l   = woT   + (size_t)l * D_MODEL * D_MODEL;
    const bf16* wu_l   = wupT  + (size_t)l * FF * D_MODEL;
    const bf16* wd_l   = wdnT  + (size_t)l * D_MODEL * FF;
    const float* bqkv_l = b_qkv  + (size_t)l * 3 * D_MODEL;
    const float* bo_l   = b_o    + (size_t)l * D_MODEL;
    const float* bu_l   = b_up   + (size_t)l * FF;
    const float* bd_l   = b_down + (size_t)l * D_MODEL;

    rmsnorm_kernel<<<S_TOK, 256, 0, stream>>>(x, g1 + (size_t)l * D_MODEL, h_bf, D_MODEL);
    gemm_bf16_async<0><<<gblk(3 * D_MODEL, S_TOK), blk, 0, stream>>>(
        h_bf, wqkv_l, bqkv_l, nullptr, qkv, S_TOK, 3 * D_MODEL, D_MODEL, 0);
    rope_pad_kernel<<<dim3(S_TOK, NH), 96, 0, stream>>>(qkv, rcos, rsin, qp, kp, vp);
    flash_attn_kernel<<<dim3(S_TOK / 64, NH), 128, 0, stream>>>(
        qp, kp, vp, rowS, rowE, flags, l, a_bf);
    gemm_bf16_async<0><<<gblk(D_MODEL, S_TOK), blk, 0, stream>>>(
        a_bf, wo_l, bo_l, x, x, S_TOK, D_MODEL, D_MODEL, 0);
    rmsnorm_kernel<<<S_TOK, 256, 0, stream>>>(x, g2 + (size_t)l * D_MODEL, h_bf, D_MODEL);
    gemm_bf16_async<1><<<gblk(FF, S_TOK), blk, 0, stream>>>(
        h_bf, wu_l, bu_l, nullptr, ff_bf, S_TOK, FF, D_MODEL, 1);
    gemm_bf16_async<0><<<gblk(D_MODEL, S_TOK), blk, 0, stream>>>(
        ff_bf, wd_l, bd_l, x, x, S_TOK, D_MODEL, FF, 0);
  }

  rmsnorm_kernel<<<S_TOK, 256, 0, stream>>>(x, final_g, h_bf, D_MODEL);
  gemm_bf16_async<1><<<gblk(MERGE_DIM, SROWS), blk, 0, stream>>>(
      h_bf, mw1T, m_b1, nullptr, ff_bf, SROWS, MERGE_DIM, MERGE_DIM, 1);
  gemm_bf16_async<0><<<gblk(OUT_DIM, SROWS), blk, 0, stream>>>(
      ff_bf, mw2T, m_b2, nullptr, y, SROWS, OUT_DIM, MERGE_DIM, 0);
  gather_rows_kernel<<<SROWS, 256, 0, stream>>>(y, rev, outp, OUT_DIM);
}
